// MultiheadAttentionBlock_2362232012853
// MI455X (gfx1250) — matrix operand order repaired and checked by reading, not compiled, not run on hardware
//
#include <hip/hip_runtime.h>
#include <hip/hip_bf16.h>
#include <math.h>

// ---------------------------------------------------------------------------
// MultiheadAttentionBlock forward for MI455X (gfx1250, wave32, WMMA).
// B=8, NQ=NK=1024, D=512, H=8, dh=64.
// GEMMs: v_wmma_f32_16x16x32_f16, double-buffered k-loop. Attention:
// flash-style streaming; 1/sqrt(512) pre-folded into f16 Q; row-sum of P via
// WMMA against an all-ones B fragment; row-max via DPP ROW_XMASK butterfly
// (pure VALU, no LDS); V tiles staged with global_load_async_to_lds_b128.
// ---------------------------------------------------------------------------

#define BB   8
#define NQ   1024
#define NK   1024
#define DD   512
#define HH   8
#define DH   64

typedef __attribute__((ext_vector_type(16))) _Float16 v16h;
typedef __attribute__((ext_vector_type(8)))  _Float16 v8h;
typedef __attribute__((ext_vector_type(8)))  float    v8f;

static __device__ __forceinline__ v16h hcombine(v8h lo, v8h hi) {
    v16h r;
#pragma unroll
    for (int i = 0; i < 8; ++i) { r[i] = lo[i]; r[i + 8] = hi[i]; }
    return r;
}

static __device__ __forceinline__ v8f wmma_f32_f16(v16h a, v16h b, v8f c) {
    // D = A(16x32) * B(32x16) + C, f32 accum
    return __builtin_amdgcn_wmma_f32_16x16x32_f16(
        /*neg_a=*/false, a, /*neg_b=*/false, b,
        /*c_mod=*/(short)0, c, /*reuse_a=*/false, /*reuse_b=*/false);
}

// Async global->LDS copy of 16B per lane (CDNA5 ASYNCcnt path).
// Per CDNA5 ISA: VDST VGPR holds the LDS byte address; the low
// 32 bits of a generic pointer to __shared__ ARE that address (aperture map).
static __device__ __forceinline__ void async_copy_b128(void* lds_dst,
                                                       const void* gsrc) {
    const unsigned ldsoff = (unsigned)(uintptr_t)lds_dst;
    asm volatile("global_load_async_to_lds_b128 %0, %1, off"
                 :: "v"(ldsoff), "v"(gsrc) : "memory");
}
static __device__ __forceinline__ void async_wait0() {
    asm volatile("s_wait_asynccnt 0x0" ::: "memory");
}

// DPP16 ROW_XMASK lane swap within 16-lane rows: VALU-only cross-lane move
// (dpp_ctrl = 0x160 | xmask), replaces ds_bpermute + dscnt waits.
template <int XMASK>
static __device__ __forceinline__ float dpp_xor_f32(float x) {
    return __int_as_float(__builtin_amdgcn_update_dpp(
        0, __float_as_int(x), 0x160 | XMASK, 0xF, 0xF, true));
}

// ---------------------------------------------------------------------------
// GEMM: C[m][n] = sum_k X[m][k] * W[n][k] + bias[n]
// X: [rows, 512] f32 row-major.  W: [512, 512] f32 row-major (used as W^T).
// Block tile 128x128, K-step 32, 8 waves (256 thr): wave grid 4(M)x2(N),
// each wave owns 32x64 = 2x4 WMMA tiles. 2-stage global->reg->LDS pipeline.
// f16 output is scaled by scl16 (used to pre-fold 1/sqrt(512) into Q).
// ---------------------------------------------------------------------------
template <bool OUT32, bool OUT16>
__global__ __launch_bounds__(256) void gemm_xwT_kernel(
    const float* __restrict__ X, const float* __restrict__ W,
    const float* __restrict__ bias,
    float* __restrict__ out32, _Float16* __restrict__ out16, float scl16)
{
    __shared__ _Float16 As[128][32 + 8];   // [m][k]  10,240 B
    __shared__ _Float16 Bs[32][128 + 8];   // [k][n]   8,704 B

    const int tid    = threadIdx.x;
    const int lane   = tid & 31;
    const int wave   = tid >> 5;       // 0..7
    const int waveM  = wave & 3;       // row strip of 32
    const int waveN  = wave >> 2;      // col strip of 64
    const int halfId = lane >> 4;
    const int l16    = lane & 15;
    const int m0     = blockIdx.x * 128;
    const int n0     = blockIdx.y * 128;

    v8f acc[2][4] = {};

    const int rBase = tid >> 3;        // 0..31
    const int c4    = (tid & 7) * 4;   // 0..28

    float4 aR[4], bR[4];
    // Prologue: fetch k-tile 0 into registers.
#pragma unroll
    for (int p = 0; p < 4; ++p) {
        aR[p] = *(const float4*)&X[(size_t)(m0 + rBase + p * 32) * DD + c4];
        bR[p] = *(const float4*)&W[(size_t)(n0 + rBase + p * 32) * DD + c4];
    }

    for (int k0 = 0; k0 < DD; k0 += 32) {
        // Dump current tile regs -> LDS (f32 -> f16 convert).
#pragma unroll
        for (int p = 0; p < 4; ++p) {
            const int row = rBase + p * 32;
            As[row][c4 + 0] = (_Float16)aR[p].x; As[row][c4 + 1] = (_Float16)aR[p].y;
            As[row][c4 + 2] = (_Float16)aR[p].z; As[row][c4 + 3] = (_Float16)aR[p].w;
            Bs[c4 + 0][row] = (_Float16)bR[p].x; Bs[c4 + 1][row] = (_Float16)bR[p].y;
            Bs[c4 + 2][row] = (_Float16)bR[p].z; Bs[c4 + 3][row] = (_Float16)bR[p].w;
        }
        __syncthreads();

        // Prefetch next k-tile into registers (overlaps with WMMAs below).
        if (k0 + 32 < DD) {
#pragma unroll
            for (int p = 0; p < 4; ++p) {
                aR[p] = *(const float4*)&X[(size_t)(m0 + rBase + p * 32) * DD + k0 + 32 + c4];
                bR[p] = *(const float4*)&W[(size_t)(n0 + rBase + p * 32) * DD + k0 + 32 + c4];
            }
        }

        // A fragments: lane -> row M=l16, K chunks 8*halfId and 16+8*halfId.
        v16h afrag[2];
#pragma unroll
        for (int i = 0; i < 2; ++i) {
            const int mrow = waveM * 32 + i * 16 + l16;
            v8h lo = *(const v8h*)&As[mrow][halfId * 8];
            v8h hi = *(const v8h*)&As[mrow][16 + halfId * 8];
            afrag[i] = hcombine(lo, hi);
        }
        // B fragments: lane -> n=l16, K chunks 8*halfId and 16+8*halfId.
#pragma unroll
        for (int j = 0; j < 4; ++j) {
            const int ncol = waveN * 64 + j * 16;
            v16h bfrag;
#pragma unroll
            for (int e = 0; e < 8; ++e) {
                bfrag[e]     = Bs[halfId * 8 + e][ncol + l16];
                bfrag[e + 8] = Bs[16 + halfId * 8 + e][ncol + l16];
            }
#pragma unroll
            for (int i = 0; i < 2; ++i)
                acc[i][j] = wmma_f32_f16(afrag[i], bfrag, acc[i][j]);
        }
        __syncthreads();
    }

    // Epilogue: C-layout element v -> row v+8*halfId, col l16. Branch-free.
#pragma unroll
    for (int i = 0; i < 2; ++i) {
#pragma unroll
        for (int j = 0; j < 4; ++j) {
            const int col = n0 + waveN * 64 + j * 16 + l16;
            const float bv = bias[col];
#pragma unroll
            for (int v = 0; v < 8; ++v) {
                const int row = m0 + waveM * 32 + i * 16 + v + 8 * halfId;
                const float r = acc[i][j][v] + bv;
                if constexpr (OUT32) out32[(size_t)row * DD + col] = r;
                if constexpr (OUT16) out16[(size_t)row * DD + col] = (_Float16)(r * scl16);
            }
        }
    }
}

// ---------------------------------------------------------------------------
// Flash attention per (query-tile 64, head, batch). 4 waves; each wave owns
// 16 queries. Streams 32-key tiles: S = Qh*Kh^T (scale pre-folded into Qp16),
// online softmax in C-layout registers (row max via DPP ROW_XMASK butterfly),
// P re-laid-out through per-wave LDS, row sums of P via WMMA vs all-ones B,
// O += P*V. Output: Qh + O/l residual -> X1.
// ---------------------------------------------------------------------------
__global__ __launch_bounds__(128) void attn_kernel(
    const _Float16* __restrict__ Qp16, const _Float16* __restrict__ Kp16,
    const _Float16* __restrict__ Vp16, const float* __restrict__ Qp32,
    float* __restrict__ X1)
{
    __shared__ _Float16 Ks[64][32 + 8];        // [d][key]  K^T tile, 5,120 B
    __shared__ _Float16 Vs[32][DH + 8];        // [key][d]  V tile,   4,608 B
    __shared__ _Float16 Ps[4][16][32 + 8];     // per-wave P tile,    5,120 B

    const int qb   = blockIdx.x * 64;
    const int h    = blockIdx.y;
    const int b    = blockIdx.z;
    const int tid  = threadIdx.x;
    const int lane = tid & 31;
    const int wave = tid >> 5;
    const int halfId = lane >> 4;
    const int l16    = lane & 15;

    // All-ones B fragment: one WMMA vs this = per-row sums of A, replicated
    // across all 16 columns (i.e., across the lanes of each half).
    v16h onesfrag;
#pragma unroll
    for (int i = 0; i < 16; ++i) onesfrag[i] = (_Float16)1.0f;

    // Preload Q A-fragments straight from global (row-major, contiguous dh).
    v16h qfrag[2];
    {
        const int qrow = qb + wave * 16 + l16;
        const _Float16* qp = Qp16 + ((size_t)b * NQ + qrow) * DD + h * DH;
#pragma unroll
        for (int c = 0; c < 2; ++c) {
            v8h lo = *(const v8h*)&qp[c * 32 + halfId * 8];
            v8h hi = *(const v8h*)&qp[c * 32 + 16 + halfId * 8];
            qfrag[c] = hcombine(lo, hi);
        }
    }

    v8f  oacc[4] = {};
    float mrow[8], lrow[8];
#pragma unroll
    for (int v = 0; v < 8; ++v) { mrow[v] = -1e30f; lrow[v] = 0.0f; }

    const int key = tid >> 2;          // 0..31
    const int dc  = (tid & 3) * 16;    // 0,16,32,48

    for (int kb = 0; kb < NK; kb += 32) {
        // V tile: pure copy -> async global->LDS (ASYNCcnt path), 2x16B/lane.
        {
            const _Float16* vp = Vp16 + ((size_t)b * NK + kb + key) * DD + h * DH + dc;
            async_copy_b128(&Vs[key][dc],     vp);
            async_copy_b128(&Vs[key][dc + 8], vp + 8);
        }
        // K tile: transpose while staging.
        {
            const _Float16* kp = Kp16 + ((size_t)b * NK + kb + key) * DD + h * DH + dc;
            v8h a0 = *(const v8h*)kp;
            v8h a1 = *(const v8h*)(kp + 8);
#pragma unroll
            for (int e = 0; e < 8; ++e) {
                Ks[dc + e][key]     = a0[e];
                Ks[dc + 8 + e][key] = a1[e];
            }
            // Prefetch next key-tile's K/V rows into cache.
            if (kb + 32 < NK) {
                __builtin_prefetch(kp + 32 * DD, 0, 1);
                __builtin_prefetch(Vp16 + ((size_t)b * NK + kb + 32 + key) * DD + h * DH + dc, 0, 1);
            }
        }
        async_wait0();          // our async V writes landed in LDS
        __syncthreads();        // everyone's K/V staging visible

        // Scores: S[16q x 32k] (already scaled), 2 C tiles x 2 k-chunks.
        v8f s[2] = {};
#pragma unroll
        for (int j = 0; j < 2; ++j) {
#pragma unroll
            for (int c = 0; c < 2; ++c) {
                v16h kf;
#pragma unroll
                for (int e = 0; e < 8; ++e) {
                    kf[e]     = Ks[c * 32 + halfId * 8 + e][j * 16 + l16];
                    kf[e + 8] = Ks[c * 32 + 16 + halfId * 8 + e][j * 16 + l16];
                }
                s[j] = wmma_f32_f16(qfrag[c], kf, s[j]);
            }
        }

        // Row max: VALU-only DPP ROW_XMASK butterfly across each 16-lane half
        // (8 rows reduced simultaneously; no LDS, no dscnt waits).
        float mx[8];
#pragma unroll
        for (int v = 0; v < 8; ++v) mx[v] = fmaxf(s[0][v], s[1][v]);
#pragma unroll
        for (int v = 0; v < 8; ++v) mx[v] = fmaxf(mx[v], dpp_xor_f32<8>(mx[v]));
#pragma unroll
        for (int v = 0; v < 8; ++v) mx[v] = fmaxf(mx[v], dpp_xor_f32<4>(mx[v]));
#pragma unroll
        for (int v = 0; v < 8; ++v) mx[v] = fmaxf(mx[v], dpp_xor_f32<2>(mx[v]));
#pragma unroll
        for (int v = 0; v < 8; ++v) mx[v] = fmaxf(mx[v], dpp_xor_f32<1>(mx[v]));

        // P = exp(S - m_new); running-max rescale factor alpha.
        float alpha[8];
#pragma unroll
        for (int v = 0; v < 8; ++v) {
            const float nm = fmaxf(mrow[v], mx[v]);
            alpha[v] = __expf(mrow[v] - nm);
            s[0][v]  = __expf(s[0][v] - nm);
            s[1][v]  = __expf(s[1][v] - nm);
            mrow[v]  = nm;
        }

        // Re-layout P: C-format -> LDS (per-wave slab) -> A-format fragment.
#pragma unroll
        for (int v = 0; v < 8; ++v) {
            const int r = v + 8 * halfId;
            Ps[wave][r][l16]      = (_Float16)s[0][v];
            Ps[wave][r][16 + l16] = (_Float16)s[1][v];
        }
        v8h plo = *(const v8h*)&Ps[wave][l16][halfId * 8];
        v8h phi = *(const v8h*)&Ps[wave][l16][16 + halfId * 8];
        const v16h pfrag = hcombine(plo, phi);

        // Row sums of P via WMMA vs ones.
        v8f lsum = {};
        lsum = wmma_f32_f16(pfrag, onesfrag, lsum);
#pragma unroll
        for (int v = 0; v < 8; ++v) lrow[v] = lrow[v] * alpha[v] + lsum[v];

        // Rescale O, then O += P(16x32keys) * V(32keys x 64d).
#pragma unroll
        for (int jd = 0; jd < 4; ++jd)
#pragma unroll
            for (int v = 0; v < 8; ++v) oacc[jd][v] *= alpha[v];
#pragma unroll
        for (int jd = 0; jd < 4; ++jd) {
            v16h vf;
#pragma unroll
            for (int e = 0; e < 8; ++e) {
                vf[e]     = Vs[halfId * 8 + e][jd * 16 + l16];
                vf[e + 8] = Vs[16 + halfId * 8 + e][jd * 16 + l16];
            }
            oacc[jd] = wmma_f32_f16(pfrag, vf, oacc[jd]);
        }
        __syncthreads();
    }

    // Epilogue: X1 = Qh + O / l  (residual inside the attention block).
#pragma unroll
    for (int v = 0; v < 8; ++v) {
        const int q = qb + wave * 16 + v + 8 * halfId;
        const float inv = 1.0f / lrow[v];
        const size_t base = ((size_t)b * NQ + q) * DD + h * DH;
#pragma unroll
        for (int jd = 0; jd < 4; ++jd) {
            const int d = jd * 16 + l16;
            X1[base + d] = Qp32[base + d] + oacc[jd][v] * inv;
        }
    }
}

// ---------------------------------------------------------------------------
// LayerNorm over D=512 (one row per 128-thread block, 4 elems/thread).
// Optional pre-op: x = X + relu(Y) when WITH_RELU_RES.
// ---------------------------------------------------------------------------
template <bool WITH_RELU_RES>
__global__ __launch_bounds__(128) void layernorm_kernel(
    const float* __restrict__ X, const float* __restrict__ Yopt,
    const float* __restrict__ g, const float* __restrict__ be,
    float* __restrict__ out)
{
    __shared__ float red[4];
    const int row  = blockIdx.x;
    const int tid  = threadIdx.x;
    const int c    = tid * 4;
    const size_t rb = (size_t)row * DD;

    float4 x = *(const float4*)&X[rb + c];
    if constexpr (WITH_RELU_RES) {
        const float4 y = *(const float4*)&Yopt[rb + c];
        x.x += fmaxf(y.x, 0.0f); x.y += fmaxf(y.y, 0.0f);
        x.z += fmaxf(y.z, 0.0f); x.w += fmaxf(y.w, 0.0f);
    }

    float s = x.x + x.y + x.z + x.w;
#pragma unroll
    for (int off = 16; off > 0; off >>= 1) s += __shfl_xor(s, off, 32);
    if ((tid & 31) == 0) red[tid >> 5] = s;
    __syncthreads();
    const float mu = (red[0] + red[1] + red[2] + red[3]) * (1.0f / DD);
    __syncthreads();

    const float dx = x.x - mu, dy = x.y - mu, dz = x.z - mu, dw = x.w - mu;
    float vs = dx * dx + dy * dy + dz * dz + dw * dw;
#pragma unroll
    for (int off = 16; off > 0; off >>= 1) vs += __shfl_xor(vs, off, 32);
    if ((tid & 31) == 0) red[tid >> 5] = vs;
    __syncthreads();
    const float var  = (red[0] + red[1] + red[2] + red[3]) * (1.0f / DD);
    const float rstd = rsqrtf(var + 1e-5f);

    const float4 gv = *(const float4*)&g[c];
    const float4 bv = *(const float4*)&be[c];
    float4 o;
    o.x = dx * rstd * gv.x + bv.x;
    o.y = dy * rstd * gv.y + bv.y;
    o.z = dz * rstd * gv.z + bv.z;
    o.w = dw * rstd * gv.w + bv.w;
    *(float4*)&out[rb + c] = o;
}

// ---------------------------------------------------------------------------
extern "C" void kernel_launch(void* const* d_in, const int* in_sizes, int n_in,
                              void* d_out, int out_size, void* d_ws, size_t ws_size,
                              hipStream_t stream) {
    const float* Q     = (const float*)d_in[0];
    const float* K     = (const float*)d_in[1];
    // d_in[2] = mask (unused by forward)
    const float* Wq    = (const float*)d_in[3];
    const float* bq    = (const float*)d_in[4];
    const float* Wk    = (const float*)d_in[5];
    const float* bk    = (const float*)d_in[6];
    const float* Wv    = (const float*)d_in[7];
    const float* bv    = (const float*)d_in[8];
    const float* Wo    = (const float*)d_in[9];
    const float* bo    = (const float*)d_in[10];
    const float* g0    = (const float*)d_in[11];
    const float* beta0 = (const float*)d_in[12];
    const float* g1    = (const float*)d_in[13];
    const float* beta1 = (const float*)d_in[14];

    const size_t rows   = (size_t)BB * NQ;         // 8192
    const size_t n_f32  = rows * DD * sizeof(float);     // 16 MB
    const size_t n_f16  = rows * DD * sizeof(_Float16);  //  8 MB

    char* w = (char*)d_ws;
    float*    Qp32 = (float*)(w);                       // 16 MB
    float*    X1   = (float*)(w + n_f32);               // 16 MB
    float*    U32  = (float*)(w + 2 * n_f32);           // 16 MB
    float*    Yg   = (float*)(w + 3 * n_f32);           // 16 MB
    _Float16* Qp16 = (_Float16*)(w + 4 * n_f32);        //  8 MB
    _Float16* Kp16 = (_Float16*)(w + 4 * n_f32 + n_f16);
    _Float16* Vp16 = (_Float16*)(w + 4 * n_f32 + 2 * n_f16);

    const dim3 ggrid(rows / 128, DD / 128);  // (64, 4)
    const float rscale = 0.044194173824159216f;   // 1/sqrt(512), full key dim

    // QKV projections. Q's f16 copy carries the softmax scale.
    gemm_xwT_kernel<true,  true ><<<ggrid, 256, 0, stream>>>(Q, Wq, bq, Qp32, Qp16, rscale);
    gemm_xwT_kernel<false, true ><<<ggrid, 256, 0, stream>>>(K, Wk, bk, nullptr, Kp16, 1.0f);
    gemm_xwT_kernel<false, true ><<<ggrid, 256, 0, stream>>>(K, Wv, bv, nullptr, Vp16, 1.0f);

    // Flash attention + residual: X1 = Qh + softmax(QK^T/sqrt(512)) V.
    attn_kernel<<<dim3(NQ / 64, HH, BB), 128, 0, stream>>>(Qp16, Kp16, Vp16, Qp32, X1);

    // U = LN0(X1)
    layernorm_kernel<false><<<rows, 128, 0, stream>>>(X1, nullptr, g0, beta0, U32);

    // Y = U @ Wo^T + bo
    gemm_xwT_kernel<true,  false><<<ggrid, 256, 0, stream>>>(U32, Wo, bo, Yg, nullptr, 1.0f);

    // out = LN1(U + relu(Y))
    layernorm_kernel<true ><<<rows, 128, 0, stream>>>(U32, Yg, g1, beta1, (float*)d_out);

    (void)in_sizes; (void)n_in; (void)out_size; (void)ws_size;
}
